// Block_cfged_88527865905336
// MI455X (gfx1250) — matrix operand order repaired and checked by reading, not compiled, not run on hardware
//
#include <hip/hip_runtime.h>
#include <hip/hip_bf16.h>

typedef __attribute__((ext_vector_type(16))) _Float16 v16h;
typedef __attribute__((ext_vector_type(8)))  float    v8f;

#define Bb    64
#define Nn    197
#define Cc    768
#define Hh    12
#define DH    64
#define NPAD  224
#define NTOK  118
#define LL    196
#define M1    (Bb*Nn)      // 12608 = 197*64
#define M2    (Bb*(NTOK+1))// 7616  = 119*64

#define WMMA_F16(a, b, c) __builtin_amdgcn_wmma_f32_16x16x32_f16(false, a, false, b, (short)0, c, false, false)

// ---------------- WMMA fragment loaders (layouts per CDNA5 ISA 7.12.2) ---------------

// A: 16x32 f16 tile from row-major source (pointer at tile origin), stride lda.
__device__ __forceinline__ v16h load_a_rm(const _Float16* A, int lda, int lane) {
    int m  = lane & 15;
    int kh = (lane >> 4) & 1;
    v16h a;
#pragma unroll
    for (int e = 0; e < 16; ++e) {
        int v = e >> 1;
        int k = ((v >> 2) << 4) + (kh << 3) + ((v & 3) << 1) + (e & 1);
        a[e] = A[m * lda + k];
    }
    return a;
}

// B: 32x16 (KxN) f16 tile where source W is row-major [N, K] (B[k][n] = W[n][k]).
__device__ __forceinline__ v16h load_b_nk(const _Float16* W, int ldw, int lane) {
    int n  = lane & 15;
    int kg = (lane >> 4) & 1;
    v16h b;
#pragma unroll
    for (int e = 0; e < 16; ++e) b[e] = W[n * ldw + ((e >> 3) << 4) + (kg << 3) + (e & 7)];
    return b;
}

// 32(M)x64(N) per-wave tile: 2 A x 4 B fragments -> 8 WMMAs per K-step,
// 12 b128 loads per 8 WMMAs (1.5 loads/WMMA); B reused across M in registers.
__device__ __forceinline__ void gemm_nk_32x64(const _Float16* __restrict__ A, int lda,
                                              const _Float16* __restrict__ W, int ldw,
                                              int K, int lane, v8f acc[8]) {
    for (int k0 = 0; k0 < K; k0 += 32) {
        __builtin_prefetch(A + k0 + 32, 0, 0);   // speculative: safe past end
        __builtin_prefetch(W + k0 + 32, 0, 0);
        v16h a0 = load_a_rm(A + k0, lda, lane);
        v16h a1 = load_a_rm(A + (long)16 * lda + k0, lda, lane);
        v16h b0 = load_b_nk(W + k0, ldw, lane);
        v16h b1 = load_b_nk(W + (long)16 * ldw + k0, ldw, lane);
        v16h b2 = load_b_nk(W + (long)32 * ldw + k0, ldw, lane);
        v16h b3 = load_b_nk(W + (long)48 * ldw + k0, ldw, lane);
        acc[0] = WMMA_F16(a0, b0, acc[0]);
        acc[1] = WMMA_F16(a0, b1, acc[1]);
        acc[2] = WMMA_F16(a0, b2, acc[2]);
        acc[3] = WMMA_F16(a0, b3, acc[3]);
        acc[4] = WMMA_F16(a1, b0, acc[4]);
        acc[5] = WMMA_F16(a1, b1, acc[5]);
        acc[6] = WMMA_F16(a1, b2, acc[6]);
        acc[7] = WMMA_F16(a1, b3, acc[7]);
    }
}

// -------------------------------- utility kernels -----------------------------------

__global__ void zero_u32_kernel(unsigned* p, long n) {
    long i = (long)blockIdx.x * blockDim.x + threadIdx.x;
    if (i < n) p[i] = 0u;
}

__global__ void cvt_f32_f16_kernel(const float* s, _Float16* d, long n) {
    long i = (long)blockIdx.x * blockDim.x + threadIdx.x;
    if (i < n) d[i] = (_Float16)s[i];
}

// LayerNorm over C=768, one block per token, write f16.
__global__ __launch_bounds__(256) void ln_kernel(const float* __restrict__ x,
                                                 const float* __restrict__ w,
                                                 const float* __restrict__ b,
                                                 _Float16* __restrict__ out) {
    long t = blockIdx.x;
    const float* row = x + t * Cc;
    __shared__ float red[256];
    float s = 0.f;
    for (int c = threadIdx.x; c < Cc; c += 256) s += row[c];
    red[threadIdx.x] = s; __syncthreads();
    for (int o = 128; o > 0; o >>= 1) { if (threadIdx.x < o) red[threadIdx.x] += red[threadIdx.x + o]; __syncthreads(); }
    float mean = red[0] * (1.f / Cc);
    __syncthreads();
    float vs = 0.f;
    for (int c = threadIdx.x; c < Cc; c += 256) { float d = row[c] - mean; vs += d * d; }
    red[threadIdx.x] = vs; __syncthreads();
    for (int o = 128; o > 0; o >>= 1) { if (threadIdx.x < o) red[threadIdx.x] += red[threadIdx.x + o]; __syncthreads(); }
    float inv = rsqrtf(red[0] * (1.f / Cc) + 1e-5f);
    for (int c = threadIdx.x; c < Cc; c += 256)
        out[t * Cc + c] = (_Float16)((row[c] - mean) * inv * w[c] + b[c]);
}

// ------------------------------------- GEMMs -----------------------------------------
// Blocks: 64 threads = 2 waves; wave w handles m0=(bx*2+w)*32, both share n0=by*64.

// qkv = lnA @ qkv_w^T; scatter q/k row-major [B,H,224,64] (q *= 1/8), v transposed [B,H,64,224].
__global__ __launch_bounds__(64) void gemm_qkv_kernel(const _Float16* __restrict__ A,
                                                      const _Float16* __restrict__ W,
                                                      _Float16* __restrict__ q,
                                                      _Float16* __restrict__ k,
                                                      _Float16* __restrict__ vT) {
    int lane = threadIdx.x & 31, wave = threadIdx.x >> 5;
    int m0 = (blockIdx.x * 2 + wave) * 32;
    int n0 = blockIdx.y * 64;
    v8f acc[8] = {};
    gemm_nk_32x64(A + (long)m0 * Cc, Cc, W + (long)n0 * Cc, Cc, Cc, lane, acc);
    int rb = (lane >> 4) * 8;
#pragma unroll
    for (int mi = 0; mi < 2; ++mi)
#pragma unroll
    for (int nt = 0; nt < 4; ++nt) {
        int col = n0 + nt * 16 + (lane & 15);
        int which = col / Cc, cc = col % Cc;
        int h = cc / DH, d = cc % DH;
#pragma unroll
        for (int vi = 0; vi < 8; ++vi) {
            int m = m0 + mi * 16 + rb + vi;
            int bb = m / Nn, n = m % Nn;
            float val = acc[mi * 4 + nt][vi];
            if (which == 0)
                q[(((long)bb * Hh + h) * NPAD + n) * DH + d] = (_Float16)(val * 0.125f);
            else if (which == 1)
                k[(((long)bb * Hh + h) * NPAD + n) * DH + d] = (_Float16)val;
            else
                vT[(((long)bb * Hh + h) * DH + d) * NPAD + n] = (_Float16)val;
        }
    }
}

// Fused attention: one wave per (b, h, 16-row tile).
__global__ __launch_bounds__(32) void attn_kernel(const _Float16* __restrict__ q,
                                                  const _Float16* __restrict__ k,
                                                  const _Float16* __restrict__ vT,
                                                  _Float16* __restrict__ ctx,
                                                  float* __restrict__ row0,
                                                  float* __restrict__ colsum) {
    __shared__ float s[16][NPAD];
    int lane = threadIdx.x;
    int mt = blockIdx.x, h = blockIdx.y, b = blockIdx.z;
    long bh = (long)b * Hh + h;
    const _Float16* qb  = q  + bh * NPAD * DH;
    const _Float16* kb  = k  + bh * NPAD * DH;
    const _Float16* vtb = vT + bh * DH * NPAD;
    int m0 = mt * 16;

    // S = (q/8) @ k^T  -> LDS ; q fragments hoisted out of the column-tile loop
    v16h aq0 = load_a_rm(qb + (long)m0 * DH + 0,  DH, lane);
    v16h aq1 = load_a_rm(qb + (long)m0 * DH + 32, DH, lane);
    for (int jt = 0; jt < 13; ++jt) {
        v8f acc = {};
        v16h b0 = load_b_nk(kb + (long)jt * 16 * DH + 0,  DH, lane);
        v16h b1 = load_b_nk(kb + (long)jt * 16 * DH + 32, DH, lane);
        acc = WMMA_F16(aq0, b0, acc);
        acc = WMMA_F16(aq1, b1, acc);
        int col = jt * 16 + (lane & 15);
        int rb  = (lane >> 4) * 8;
#pragma unroll
        for (int vi = 0; vi < 8; ++vi) s[rb + vi][col] = acc[vi];
    }
    for (int i = lane; i < 16 * 16; i += 32) s[i >> 4][208 + (i & 15)] = 0.f;
    __syncthreads();

    // softmax over j<197 per row; zero padding
    for (int r = 0; r < 16; ++r) {
        float mx = -1e30f;
        for (int j = lane; j < Nn; j += 32) mx = fmaxf(mx, s[r][j]);
        for (int o = 16; o > 0; o >>= 1) mx = fmaxf(mx, __shfl_xor(mx, o, 32));
        float sum = 0.f;
        for (int j = lane; j < Nn; j += 32) { float e = expf(s[r][j] - mx); s[r][j] = e; sum += e; }
        for (int o = 16; o > 0; o >>= 1) sum += __shfl_xor(sum, o, 32);
        float inv = 1.f / sum;
        for (int j = lane; j < NPAD; j += 32) s[r][j] = (j < Nn) ? s[r][j] * inv : 0.f;
    }
    __syncthreads();

    // selector side-outputs
    if (mt == 0)
        for (int j = lane; j < NPAD; j += 32) row0[bh * NPAD + j] = s[0][j];
    for (int j = lane; j < NPAD; j += 32) {
        float p = 0.f;
        for (int r = 0; r < 16; ++r) {
            int gi = m0 + r;
            if (gi >= 1 && gi < Nn) p += s[r][j];
        }
        atomicAdd(&colsum[bh * NPAD + j], p);
    }
    __syncthreads();

    // out = attn @ v : A fragment (LDS f32->f16) built once per K-step, 4 accumulators
    {
        v8f acc0 = {}, acc1 = {}, acc2 = {}, acc3 = {};
        int m = lane & 15, kh = lane >> 4;
        for (int k0 = 0; k0 < NPAD; k0 += 32) {
            v16h a;
#pragma unroll
            for (int e = 0; e < 16; ++e) {
                int vv = e >> 1;
                int kk = ((vv >> 2) << 4) + (kh << 3) + ((vv & 3) << 1) + (e & 1);
                a[e] = (_Float16)s[m][k0 + kk];
            }
            v16h b0 = load_b_nk(vtb + (long)0  * NPAD + k0, NPAD, lane);
            v16h b1 = load_b_nk(vtb + (long)16 * NPAD + k0, NPAD, lane);
            v16h b2 = load_b_nk(vtb + (long)32 * NPAD + k0, NPAD, lane);
            v16h b3 = load_b_nk(vtb + (long)48 * NPAD + k0, NPAD, lane);
            acc0 = WMMA_F16(a, b0, acc0);
            acc1 = WMMA_F16(a, b1, acc1);
            acc2 = WMMA_F16(a, b2, acc2);
            acc3 = WMMA_F16(a, b3, acc3);
        }
        int rb = (lane >> 4) * 8;
        v8f accs[4] = {acc0, acc1, acc2, acc3};
#pragma unroll
        for (int nt = 0; nt < 4; ++nt) {
            int col = nt * 16 + (lane & 15);
#pragma unroll
            for (int vi = 0; vi < 8; ++vi) {
                int n = m0 + rb + vi;
                if (n < Nn)
                    ctx[((long)b * Nn + n) * Cc + h * DH + col] = (_Float16)accs[nt][vi];
            }
        }
    }
}

// xs = gamma1*(ctx @ proj_w^T + proj_b) + x
__global__ __launch_bounds__(64) void gemm_proj_kernel(const _Float16* __restrict__ A,
                                                       const _Float16* __restrict__ W,
                                                       const float* __restrict__ bias,
                                                       const float* __restrict__ g1,
                                                       const float* __restrict__ x,
                                                       float* __restrict__ xs) {
    int lane = threadIdx.x & 31, wave = threadIdx.x >> 5;
    int m0 = (blockIdx.x * 2 + wave) * 32;
    int n0 = blockIdx.y * 64;
    v8f acc[8] = {};
    gemm_nk_32x64(A + (long)m0 * Cc, Cc, W + (long)n0 * Cc, Cc, Cc, lane, acc);
    int rb = (lane >> 4) * 8;
#pragma unroll
    for (int mi = 0; mi < 2; ++mi)
#pragma unroll
    for (int nt = 0; nt < 4; ++nt) {
        int col = n0 + nt * 16 + (lane & 15);
#pragma unroll
        for (int vi = 0; vi < 8; ++vi) {
            long m = m0 + mi * 16 + rb + vi;
            xs[m * Cc + col] = g1[col] * (acc[mi * 4 + nt][vi] + bias[col]) + x[m * Cc + col];
        }
    }
}

// h1 = gelu(lnB @ fc1_w^T + fc1_b)  (exact gelu), store f16
__global__ __launch_bounds__(64) void gemm_fc1_kernel(const _Float16* __restrict__ A,
                                                      const _Float16* __restrict__ W,
                                                      const float* __restrict__ bias,
                                                      _Float16* __restrict__ h1) {
    int lane = threadIdx.x & 31, wave = threadIdx.x >> 5;
    int m0 = (blockIdx.x * 2 + wave) * 32;
    int n0 = blockIdx.y * 64;
    v8f acc[8] = {};
    gemm_nk_32x64(A + (long)m0 * Cc, Cc, W + (long)n0 * Cc, Cc, Cc, lane, acc);
    int rb = (lane >> 4) * 8;
#pragma unroll
    for (int mi = 0; mi < 2; ++mi)
#pragma unroll
    for (int nt = 0; nt < 4; ++nt) {
        int col = n0 + nt * 16 + (lane & 15);
#pragma unroll
        for (int vi = 0; vi < 8; ++vi) {
            long m = m0 + mi * 16 + rb + vi;
            float t = acc[mi * 4 + nt][vi] + bias[col];
            float g = 0.5f * t * (1.f + erff(t * 0.70710678118654752f));
            h1[m * (4 * Cc) + col] = (_Float16)g;
        }
    }
}

// out = gamma2*(h1 @ fc2_w^T + fc2_b) + xsel
__global__ __launch_bounds__(64) void gemm_fc2_kernel(const _Float16* __restrict__ A,
                                                      const _Float16* __restrict__ W,
                                                      const float* __restrict__ bias,
                                                      const float* __restrict__ g2,
                                                      const float* __restrict__ xsel,
                                                      float* __restrict__ out) {
    int lane = threadIdx.x & 31, wave = threadIdx.x >> 5;
    int m0 = (blockIdx.x * 2 + wave) * 32;
    int n0 = blockIdx.y * 64;
    v8f acc[8] = {};
    gemm_nk_32x64(A + (long)m0 * (4 * Cc), 4 * Cc, W + (long)n0 * (4 * Cc), 4 * Cc, 4 * Cc, lane, acc);
    int rb = (lane >> 4) * 8;
#pragma unroll
    for (int mi = 0; mi < 2; ++mi)
#pragma unroll
    for (int nt = 0; nt < 4; ++nt) {
        int col = n0 + nt * 16 + (lane & 15);
#pragma unroll
        for (int vi = 0; vi < 8; ++vi) {
            long m = m0 + mi * 16 + rb + vi;
            out[m * Cc + col] = g2[col] * (acc[mi * 4 + nt][vi] + bias[col]) + xsel[m * Cc + col];
        }
    }
}

// ------------------------------- selector kernels ------------------------------------

__global__ void token_attn_kernel(const float* __restrict__ row0,
                                  const float* __restrict__ colsum,
                                  float* __restrict__ tattn) {
    int b = blockIdx.x, t = threadIdx.x;
    if (t >= LL) return;
    float acc = 0.f;
    for (int h = 0; h < Hh; ++h) {
        long bh = (long)b * Hh + h;
        float cls_aw = colsum[bh * NPAD + 0] * (1.f / LL);   // mean_{i>=1} attn[i,0]
        float img_aw = row0[bh * NPAD + 0];                  // attn[0,0]
        acc += row0[bh * NPAD + 1 + t] * cls_aw
             + colsum[bh * NPAD + 1 + t] * (1.f / LL) * img_aw;
    }
    tattn[b * LL + t] = acc * (1.f / Hh);
}

// G = IFFT(high-pass notch kernel): G[d] = (1/L) sum_f kern[f] e^{2pi i f d / L}
__global__ void gcoef_kernel(float* __restrict__ G) {
    int d = threadIdx.x;
    if (d >= LL) return;
    const float L = (float)LL;
    float d0 = (L * 0.15f * 0.5f); d0 = d0 * d0;
    float m0 = (L - 1.f) * 0.5f;
    float re = 0.f, im = 0.f;
    for (int f = 0; f < LL; ++f) {
        float g = (float)f;
        float kern = 1.f - expf(-((g - m0) * (g - m0)) / (2.f * d0));
        float ang = 6.28318530717958647692f * (float)f * (float)d / L;
        re += kern * cosf(ang);
        im += kern * sinf(ang);
    }
    G[d] = re / L; G[LL + d] = im / L;
}

// tok = circular conv(v, G); accumulate sum_c |tok| and sum_{c,t} |v|
__global__ __launch_bounds__(256) void lfe_kernel(const float* __restrict__ xs,
                                                  const float* __restrict__ G,
                                                  float* __restrict__ lfe,
                                                  float* __restrict__ denom) {
    int c = blockIdx.x, b = blockIdx.y;
    __shared__ float v[LL], gre[LL], gim[LL], red[256];
    int t = threadIdx.x;
    if (t < LL) {
        v[t]   = xs[((long)b * Nn + 1 + t) * Cc + c];
        gre[t] = G[t];
        gim[t] = G[LL + t];
    }
    __syncthreads();
    float absv = 0.f;
    if (t < LL) {
        absv = fabsf(v[t]);
        float re = 0.f, im = 0.f;
        for (int s = 0; s < LL; ++s) {
            int d = t - s; if (d < 0) d += LL;
            re += v[s] * gre[d];
            im += v[s] * gim[d];
        }
        atomicAdd(&lfe[b * LL + t], sqrtf(re * re + im * im));
    }
    red[t] = absv; __syncthreads();
    for (int o = 128; o > 0; o >>= 1) { if (t < o) red[t] += red[t + o]; __syncthreads(); }
    if (t == 0) atomicAdd(&denom[b], red[0]);
}

// scores + serial top-118 (first-index tie-break, matches lax.top_k)
__global__ __launch_bounds__(256) void topk_kernel(const float* __restrict__ lfe,
                                                   const float* __restrict__ denom,
                                                   const float* __restrict__ tattn,
                                                   int* __restrict__ idx) {
    int b = blockIdx.x, t = threadIdx.x;
    __shared__ float sc[LL];
    if (t < LL) sc[t] = lfe[b * LL + t] / (denom[b] * (1.f / LL)) * tattn[b * LL + t];
    __syncthreads();
    if (t == 0) {
        for (int i = 0; i < NTOK; ++i) {
            float best = -3.4e38f; int bj = 0;
            for (int j = 0; j < LL; ++j) if (sc[j] > best) { best = sc[j]; bj = j; }
            idx[b * NTOK + i] = bj;
            sc[bj] = -3.4e38f;
        }
    }
}

__global__ void gather_kernel(const float* __restrict__ xs, const int* __restrict__ idx,
                              float* __restrict__ xsel) {
    int r = blockIdx.x, b = blockIdx.y;
    int src = (r == 0) ? 0 : 1 + idx[b * NTOK + (r - 1)];
    const float* s = xs + ((long)b * Nn + src) * Cc;
    float* d = xsel + ((long)b * (NTOK + 1) + r) * Cc;
    for (int c = threadIdx.x; c < Cc; c += blockDim.x) d[c] = s[c];
}

// --------------------------------- launch --------------------------------------------

extern "C" void kernel_launch(void* const* d_in, const int* in_sizes, int n_in,
                              void* d_out, int out_size, void* d_ws, size_t ws_size,
                              hipStream_t stream) {
    (void)in_sizes; (void)n_in; (void)out_size; (void)ws_size;
    const float* x      = (const float*)d_in[0];
    const float* ln1_w  = (const float*)d_in[1];
    const float* ln1_b  = (const float*)d_in[2];
    const float* qkv_w  = (const float*)d_in[3];
    const float* proj_w = (const float*)d_in[4];
    const float* proj_b = (const float*)d_in[5];
    const float* gamma1 = (const float*)d_in[6];
    const float* ln2_w  = (const float*)d_in[7];
    const float* ln2_b  = (const float*)d_in[8];
    const float* fc1_w  = (const float*)d_in[9];
    const float* fc1_b  = (const float*)d_in[10];
    const float* fc2_w  = (const float*)d_in[11];
    const float* fc2_b  = (const float*)d_in[12];
    const float* gamma2 = (const float*)d_in[13];
    float* out = (float*)d_out;

    char* ws = (char*)d_ws;
    size_t off = 0;
    auto alloc = [&](size_t bytes) { size_t o = off; off += (bytes + 255) & ~(size_t)255; return o; };

    _Float16* wqkv  = (_Float16*)(ws + alloc((size_t)3 * Cc * Cc * 2));
    _Float16* wproj = (_Float16*)(ws + alloc((size_t)Cc * Cc * 2));
    _Float16* wfc1  = (_Float16*)(ws + alloc((size_t)4 * Cc * Cc * 2));
    _Float16* wfc2  = (_Float16*)(ws + alloc((size_t)4 * Cc * Cc * 2));
    _Float16* lnA   = (_Float16*)(ws + alloc((size_t)M1 * Cc * 2));
    _Float16* ctx   = (_Float16*)(ws + alloc((size_t)M1 * Cc * 2));
    float*    xs    = (float*)   (ws + alloc((size_t)M1 * Cc * 4));
    float*    tattn = (float*)   (ws + alloc((size_t)Bb * LL * 4));
    float*    G     = (float*)   (ws + alloc((size_t)2 * LL * 4));
    int*      idx   = (int*)     (ws + alloc((size_t)Bb * NTOK * 4));
    float*    xsel  = (float*)   (ws + alloc((size_t)M2 * Cc * 4));
    _Float16* lnB   = (_Float16*)(ws + alloc((size_t)M2 * Cc * 2));
    _Float16* h1    = (_Float16*)(ws + alloc((size_t)M2 * 4 * Cc * 2));
    // zero-initialized region (padded q/k/vT + accumulators), contiguous:
    size_t zoff = off;
    _Float16* qb    = (_Float16*)(ws + alloc((size_t)Bb * Hh * NPAD * DH * 2));
    _Float16* kb    = (_Float16*)(ws + alloc((size_t)Bb * Hh * NPAD * DH * 2));
    _Float16* vT    = (_Float16*)(ws + alloc((size_t)Bb * Hh * DH * NPAD * 2));
    float*    row0  = (float*)   (ws + alloc((size_t)Bb * Hh * NPAD * 4));
    float*    csum  = (float*)   (ws + alloc((size_t)Bb * Hh * NPAD * 4));
    float*    lfe   = (float*)   (ws + alloc((size_t)Bb * LL * 4));
    float*    denom = (float*)   (ws + alloc((size_t)Bb * 4));
    long zn = (long)((off - zoff) / 4);

    zero_u32_kernel<<<(unsigned)((zn + 255) / 256), 256, 0, stream>>>((unsigned*)(ws + zoff), zn);

    cvt_f32_f16_kernel<<<(3 * Cc * Cc + 255) / 256, 256, 0, stream>>>(qkv_w,  wqkv,  (long)3 * Cc * Cc);
    cvt_f32_f16_kernel<<<(Cc * Cc + 255) / 256,     256, 0, stream>>>(proj_w, wproj, (long)Cc * Cc);
    cvt_f32_f16_kernel<<<(4 * Cc * Cc + 255) / 256, 256, 0, stream>>>(fc1_w,  wfc1,  (long)4 * Cc * Cc);
    cvt_f32_f16_kernel<<<(4 * Cc * Cc + 255) / 256, 256, 0, stream>>>(fc2_w,  wfc2,  (long)4 * Cc * Cc);

    ln_kernel<<<M1, 256, 0, stream>>>(x, ln1_w, ln1_b, lnA);
    gemm_qkv_kernel<<<dim3(M1 / 64, (3 * Cc) / 64), 64, 0, stream>>>(lnA, wqkv, qb, kb, vT);
    attn_kernel<<<dim3(13, Hh, Bb), 32, 0, stream>>>(qb, kb, vT, ctx, row0, csum);
    gemm_proj_kernel<<<dim3(M1 / 64, Cc / 64), 64, 0, stream>>>(ctx, wproj, proj_b, gamma1, x, xs);

    token_attn_kernel<<<Bb, 256, 0, stream>>>(row0, csum, tattn);
    gcoef_kernel<<<1, 256, 0, stream>>>(G);
    lfe_kernel<<<dim3(Cc, Bb), 256, 0, stream>>>(xs, G, lfe, denom);
    topk_kernel<<<Bb, 256, 0, stream>>>(lfe, denom, tattn, idx);
    gather_kernel<<<dim3(NTOK + 1, Bb), 256, 0, stream>>>(xs, idx, xsel);

    ln_kernel<<<M2, 256, 0, stream>>>(xsel, ln2_w, ln2_b, lnB);
    gemm_fc1_kernel<<<dim3(M2 / 64, (4 * Cc) / 64), 64, 0, stream>>>(lnB, wfc1, fc1_b, h1);
    gemm_fc2_kernel<<<dim3(M2 / 64, Cc / 64), 64, 0, stream>>>(h1, wfc2, fc2_b, gamma2, xsel, out);
}
